// LSTM_decoder_new_76562087018910
// MI455X (gfx1250) — matrix operand order repaired and checked by reading, not compiled, not run on hardware
//
#include <hip/hip_runtime.h>
#include <hip/hip_bf16.h>

// ---------------------------------------------------------------------------
// LSTM decoder for MI455X (gfx1250, wave32, WMMA bf16 16x16x32, f32 accum).
// Batch-parallel persistent kernel: 64 WGs x 16 batch rows, S=256 steps
// fully on-chip. Weights pre-packed once into WMMA fragment layouts and
// streamed from the (L2-resident) packed blobs every step. Opaque pointer
// recurrences stop LICM from hoisting the loop-invariant weight loads
// (which previously spilled 2KB/lane to scratch).
// ---------------------------------------------------------------------------

typedef __attribute__((ext_vector_type(16))) __bf16        v16bf;
typedef __attribute__((ext_vector_type(8)))  float         v8f;
typedef __attribute__((ext_vector_type(8)))  unsigned int  v8u;

#define ZDIM  64
#define NFEAT 32
#define HDIM  256
#define SLEN  256
#define BATCH 1024

// ---- helpers --------------------------------------------------------------

__device__ __forceinline__ unsigned short f32_to_bf16(float f) {
    unsigned int u = __float_as_uint(f);
    unsigned int r = u + 0x7FFFu + ((u >> 16) & 1u);   // round to nearest even
    return (unsigned short)(r >> 16);
}

union FragCvt { v8u u; v16bf b; };

// 32B fragment load (works for global and LDS pointers)
__device__ __forceinline__ v16bf load_frag(const unsigned short* p) {
    const uint4* q = (const uint4*)p;
    uint4 a = q[0], c = q[1];
    FragCvt fc;
    fc.u[0] = a.x; fc.u[1] = a.y; fc.u[2] = a.z; fc.u[3] = a.w;
    fc.u[4] = c.x; fc.u[5] = c.y; fc.u[6] = c.z; fc.u[7] = c.w;
    return fc.b;
}

__device__ __forceinline__ v8f splat8(float x) {
    v8f v;
#pragma unroll
    for (int i = 0; i < 8; ++i) v[i] = x;
    return v;
}

__device__ __forceinline__ float sigmoidf_(float x) {
    return 1.0f / (1.0f + __expf(-x));
}

// ---- weight pre-pack kernels ----------------------------------------------
// B-fragment layout (16-bit B-matrix KxN=32x16, ISA 7.12.2):
//   lane = n_local + 16*((k_local>>3)&1);  idx(0..15) = 8*(k_local>>4) + (k_local&7)
//   blob[((frag)*32 + lane)*16 + idx]
// frag element: n = nt*16 + (lane&15), k = kc*32 + 8*(lane>>4) + 16*(idx>>3) + (idx&7)

__global__ void pack_whh(const float* __restrict__ W, unsigned short* __restrict__ out) {
    int tid = blockIdx.x * 256 + threadIdx.x;      // 64*8*32*16 = 262144
    int idx = tid & 15, lane = (tid >> 4) & 31, kc = (tid >> 9) & 7, nt = tid >> 12;
    int n = nt * 16 + (lane & 15);                 // 0..1023 (4H rows)
    int k = kc * 32 + ((lane >> 4) << 3) + ((idx >> 3) << 4) + (idx & 7);    // 0..255  (H cols)
    out[tid] = f32_to_bf16(W[n * HDIM + k]);
}

__global__ void pack_wih(const float* __restrict__ W, unsigned short* __restrict__ out) {
    int tid = blockIdx.x * 256 + threadIdx.x;      // 64*32*16 = 32768
    int idx = tid & 15, lane = (tid >> 4) & 31, nt = tid >> 9;
    int n = nt * 16 + (lane & 15);                 // 0..1023
    int k = ((lane >> 4) << 3) + ((idx >> 3) << 4) + (idx & 7);              // 0..31 (NF)
    out[tid] = f32_to_bf16(W[n * NFEAT + k]);
}

__global__ void pack_wout(const float* __restrict__ W, unsigned short* __restrict__ out) {
    int tid = blockIdx.x * 256 + threadIdx.x;      // 2*8*32*16 = 8192
    int idx = tid & 15, lane = (tid >> 4) & 31, kc = (tid >> 9) & 7, nt = tid >> 12;
    int n = nt * 16 + (lane & 15);                 // 0..31 (NF rows)
    int k = kc * 32 + ((lane >> 4) << 3) + ((idx >> 3) << 4) + (idx & 7);    // 0..255 (H)
    out[tid] = f32_to_bf16(W[n * HDIM + k]);
}

// W_sig: (NF, H*S); B-matrix for step s: B[k=hh][n=nf] = W_sig[nf, hh*S + s]
__global__ void pack_wsig(const float* __restrict__ W, unsigned short* __restrict__ out) {
    int tid = blockIdx.x * 256 + threadIdx.x;      // 256*2*8*32*16 = 2097152
    int idx = tid & 15, lane = (tid >> 4) & 31, kc = (tid >> 9) & 7;
    int nt = (tid >> 12) & 1, s = tid >> 13;
    int nf = nt * 16 + (lane & 15);
    int hh = kc * 32 + ((lane >> 4) << 3) + ((idx >> 3) << 4) + (idx & 7);
    out[tid] = f32_to_bf16(W[(size_t)nf * (HDIM * SLEN) + (size_t)hh * SLEN + s]);
}

// ---- persistent LSTM kernel ------------------------------------------------
// A-fragment layout (16-bit A-matrix MxK=16x32, ISA 7.12.2):
//   lane = m + 16*((k>>3)&1); idx = 8*((k>>4)&1) + (k&7); kc = k>>5

__global__ __launch_bounds__(256, 1) void lstm_persist(
    const float* __restrict__ z,     const float* __restrict__ W_lat,
    const float* __restrict__ b_lat, const float* __restrict__ b_ih,
    const float* __restrict__ b_hh,  const float* __restrict__ b_out,
    const float* __restrict__ b_sig,
    const unsigned short* __restrict__ whhF, const unsigned short* __restrict__ wihF,
    const unsigned short* __restrict__ woutF, const unsigned short* __restrict__ wsigF,
    float* __restrict__ mu_out, float* __restrict__ sig_out)
{
    __shared__ __align__(32) unsigned short hA [8 * 32 * 16];  // h_s,  bf16 A-frags (8 KB)
    __shared__ __align__(32) unsigned short thA[8 * 32 * 16];  // tanh(h_s)        (8 KB)
    __shared__ __align__(32) unsigned short xA [32 * 16];      // x_s, K=32        (1 KB)

    const int tid  = threadIdx.x;
    const int w    = tid >> 5;        // wave 0..7
    const int lane = tid & 31;
    const int hi   = lane >> 4;       // high half of wave
    const int ln   = lane & 15;
    const int b0   = blockIdx.x * 16; // batch tile base

    // fused bias for the 8 N-tiles this wave owns (nt = w + 8j)
    float bias[8];
#pragma unroll
    for (int j = 0; j < 8; ++j) {
        int n = (w + 8 * j) * 16 + ln;
        bias[j] = b_ih[n] + b_hh[n];
    }

    // ---- init: h,c = z @ W_lat.T + b_lat (tiny K=64 GEMM, scalar f32) ----
    float c[2][8];
#pragma unroll
    for (int tt = 0; tt < 2; ++tt) {
        int n = (w + 8 * tt) * 16 + ln;            // hidden column 0..255
        for (int e = 0; e < 8; ++e) {
            int m = e + 8 * hi;
            const float* zb = z + (size_t)(b0 + m) * ZDIM;
            const float* wh = W_lat + (size_t)n * ZDIM;
            const float* wc = W_lat + (size_t)(n + HDIM) * ZDIM;
            float hv = b_lat[n], cv = b_lat[n + HDIM];
            for (int k = 0; k < ZDIM; ++k) { float zk = zb[k]; hv += zk * wh[k]; cv += zk * wc[k]; }
            c[tt][e] = cv;
            int kc = n >> 5;
            int lp = m + 16 * ((n >> 3) & 1);
            int ix = 8 * ((n >> 4) & 1) + (n & 7);
            hA [(kc * 32 + lp) * 16 + ix] = f32_to_bf16(hv);
            thA[(kc * 32 + lp) * 16 + ix] = f32_to_bf16(tanhf(hv));
        }
    }
    if (w == 0) {
#pragma unroll
        for (int i = 0; i < 16; ++i) xA[lane * 16 + i] = 0;   // x_0 = 0
    }
    v8f sacc = splat8(0.0f);                                   // sigma logits

    // Opaque zero: stops LICM from proving the weight-fragment addresses
    // loop-invariant (hoisting them spills ~2KB/lane to scratch).
    int zero;
    asm volatile("s_mov_b32 %0, 0" : "=s"(zero));
    const unsigned short* whh_p  = whhF  + (size_t)w * 512 * 8;   // this wave's nt = w
    const unsigned short* wih_p  = wihF  + (size_t)w * 512;
    const unsigned short* wout_p = woutF;

    __syncthreads();

    // ---- S sequential steps, all on-chip ----
    for (int s = 0; s < SLEN; ++s) {
        // A fragments: h_s (8 k-chunks) and x_s (1 chunk)
        v16bf a[8];
#pragma unroll
        for (int kc = 0; kc < 8; ++kc) a[kc] = load_frag(&hA[(kc * 32 + lane) * 16]);
        v16bf ax = load_frag(&xA[lane * 16]);
        __syncthreads();   // everyone has read h/x before they get overwritten

        // g = x@W_ih.T + h@W_hh.T + bias : 8 N-tiles/wave, 9 WMMA each
        // wave w owns nt = w + 8j  ->  byte offset (8j)*8 fragments from whh_p
        v8f g[8];
#pragma unroll
        for (int j = 0; j < 8; ++j) {
            v8f acc = splat8(bias[j]);
            const unsigned short* pb = whh_p + (size_t)(8 * j) * 8 * 512 + lane * 16;
#pragma unroll
            for (int kc = 0; kc < 8; ++kc) {
                v16bf bfr = load_frag(pb + kc * 512);
                acc = __builtin_amdgcn_wmma_f32_16x16x32_bf16(
                          false, a[kc], false, bfr, (short)0, acc, false, false);
            }
            v16bf bix = load_frag(wih_p + (size_t)(8 * j) * 512 + lane * 16);
            acc = __builtin_amdgcn_wmma_f32_16x16x32_bf16(
                      false, ax, false, bix, (short)0, acc, false, false);
            g[j] = acc;
        }

        // gate math: i/f/gg/o for the SAME columns live in the same lanes
        // (nt = w+8j: j=0,1 -> i ; 2,3 -> f ; 4,5 -> gg ; 6,7 -> o)
#pragma unroll
        for (int tt = 0; tt < 2; ++tt) {
            int n = (w + 8 * tt) * 16 + ln;
            int kc = n >> 5;
            int lpb = 16 * ((n >> 3) & 1);
            int ix = 8 * ((n >> 4) & 1) + (n & 7);
#pragma unroll
            for (int e = 0; e < 8; ++e) {
                float si = sigmoidf_(g[0 + tt][e]);
                float sf = sigmoidf_(g[2 + tt][e]);
                float gv = tanhf    (g[4 + tt][e]);
                float so = sigmoidf_(g[6 + tt][e]);
                float cn = sf * c[tt][e] + si * gv;
                c[tt][e] = cn;
                float hn = so * tanhf(cn);
                int m  = e + 8 * hi;
                int lp = m + lpb;
                hA [(kc * 32 + lp) * 16 + ix] = f32_to_bf16(hn);
                thA[(kc * 32 + lp) * 16 + ix] = f32_to_bf16(tanhf(hn));
            }
        }
        __syncthreads();   // new h_s visible to all waves

        if (w < 2) {
            // x_{s+1} = tanh(h_s) @ W_out.T + b_out ; mu[:, :, s] = x_{s+1}
            v8f xacc = splat8(b_out[w * 16 + ln]);
#pragma unroll
            for (int kc = 0; kc < 8; ++kc) {
                v16bf af = load_frag(&thA[(kc * 32 + lane) * 16]);
                v16bf bf = load_frag(wout_p + (size_t)(w * 8 + kc) * 512 + lane * 16);
                xacc = __builtin_amdgcn_wmma_f32_16x16x32_bf16(
                           false, af, false, bf, (short)0, xacc, false, false);
            }
            int nf = w * 16 + ln;
#pragma unroll
            for (int e = 0; e < 8; ++e) {
                int m = e + 8 * hi;
                mu_out[((size_t)(b0 + m) * NFEAT + nf) * SLEN + s] = xacc[e];
                int lp = m + 16 * ((nf >> 3) & 1);
                int ix = 8 * ((nf >> 4) & 1) + (nf & 7);
                xA[lp * 16 + ix] = f32_to_bf16(xacc[e]);
            }
        } else if (w < 4) {
            // sigma logits += h_s @ W_sig[:, hh*S + s].T  (L2-resident fragments)
            const unsigned short* ps =
                wsigF + ((size_t)(s * 2 + (w - 2)) * 8) * 512 + lane * 16;
            __builtin_prefetch(ps + 2 * 8 * 512, 0, 0);   // next step's fragments
#pragma unroll
            for (int kc = 0; kc < 8; ++kc) {
                v16bf af = load_frag(&hA[(kc * 32 + lane) * 16]);
                v16bf bf = load_frag(ps + kc * 512);
                sacc = __builtin_amdgcn_wmma_f32_16x16x32_bf16(
                           false, af, false, bf, (short)0, sacc, false, false);
            }
        }
        __syncthreads();   // x_{s+1} ready for next iteration

        // opaque no-op pointer advance: defeats hoisting of weight loads
        whh_p  += zero;
        wih_p  += zero;
        wout_p += zero;
    }

    // ---- sigma = softplus(logits + b_sig) ----
    if (w == 2 || w == 3) {
        int nf = (w - 2) * 16 + ln;
        float bs = b_sig[nf];
#pragma unroll
        for (int e = 0; e < 8; ++e) {
            int m = e + 8 * hi;
            float v = sacc[e] + bs;
            float sp = (v > 20.0f) ? v : log1pf(__expf(v));
            sig_out[(size_t)(b0 + m) * NFEAT + nf] = sp;
        }
    }
}

// ---------------------------------------------------------------------------

extern "C" void kernel_launch(void* const* d_in, const int* in_sizes, int n_in,
                              void* d_out, int out_size, void* d_ws, size_t ws_size,
                              hipStream_t stream) {
    const float* z     = (const float*)d_in[0];
    const float* W_lat = (const float*)d_in[1];
    const float* b_lat = (const float*)d_in[2];
    const float* W_ih  = (const float*)d_in[3];
    const float* b_ih  = (const float*)d_in[4];
    const float* W_hh  = (const float*)d_in[5];
    const float* b_hh  = (const float*)d_in[6];
    const float* W_out = (const float*)d_in[7];
    const float* b_out = (const float*)d_in[8];
    const float* W_sig = (const float*)d_in[9];
    const float* b_sig = (const float*)d_in[10];

    unsigned short* whhF  = (unsigned short*)d_ws;   // 262144 elems (512 KB)
    unsigned short* wihF  = whhF  + 262144;          //  32768 elems ( 64 KB)
    unsigned short* woutF = wihF  + 32768;           //   8192 elems ( 16 KB)
    unsigned short* wsigF = woutF + 8192;            // 2097152 elems ( 4 MB)

    pack_whh <<<1024, 256, 0, stream>>>(W_hh,  whhF);
    pack_wih <<<128,  256, 0, stream>>>(W_ih,  wihF);
    pack_wout<<<32,   256, 0, stream>>>(W_out, woutF);
    pack_wsig<<<8192, 256, 0, stream>>>(W_sig, wsigF);

    float* mu = (float*)d_out;
    float* sg = mu + (size_t)BATCH * NFEAT * SLEN;
    lstm_persist<<<64, 256, 0, stream>>>(z, W_lat, b_lat, b_ih, b_hh, b_out, b_sig,
                                         whhF, wihF, woutF, wsigF, mu, sg);
}
